// GraphAttentionNetwork_55602646614233
// MI455X (gfx1250) — compile-verified, matrix operand order checked by reading, not run on hardware
//
#include <hip/hip_runtime.h>

typedef __bf16 bf16_t;
typedef bf16_t bf16x16 __attribute__((ext_vector_type(16)));
typedef bf16_t bf16x8  __attribute__((ext_vector_type(8)));
typedef bf16_t bf16x4  __attribute__((ext_vector_type(4)));
typedef float  f32x8   __attribute__((ext_vector_type(8)));
typedef int    v4i_gnu __attribute__((vector_size(16)));   // matches builtin param pointee

#define NNEI 50
#define HDIM 128
#define MPAD 64

// workspace layout (elements): W_user bf16 [0,16384) | W_news [16384,32768) | W_sem [32768,81920)
// then floats: per-aggregation [bias(128), q(128)] blocks: user, news, sem
#define WS_W_ELEMS 81920

#if __has_builtin(__builtin_amdgcn_tanhf)
#define TANH_F32(x) __builtin_amdgcn_tanhf(x)
#else
#define TANH_F32(x) tanhf(x)
#endif

// ---------------- one-time weight prep: f32 -> bf16 W, fused bias ----------------
__global__ __launch_bounds__(256)
void prep_weights(const float* __restrict__ Wu, const float* __restrict__ bu,
                  const float* __restrict__ uu, const float* __restrict__ qu,
                  const float* __restrict__ Wn, const float* __restrict__ bn,
                  const float* __restrict__ un, const float* __restrict__ qn,
                  const float* __restrict__ Ws, const float* __restrict__ bs,
                  const float* __restrict__ us, const float* __restrict__ qs,
                  bf16_t* __restrict__ wsW, float* __restrict__ wsV)
{
    int idx = blockIdx.x * 256 + threadIdx.x;
    if (idx < 16384)      wsW[idx] = (bf16_t)Wu[idx];
    else if (idx < 32768) wsW[idx] = (bf16_t)Wn[idx - 16384];
    else if (idx < WS_W_ELEMS) wsW[idx] = (bf16_t)Ws[idx - 32768];
    if (idx < HDIM) {
        wsV[idx]       = bu[idx] + uu[idx];  wsV[128 + idx] = qu[idx];
        wsV[256 + idx] = bn[idx] + un[idx];  wsV[384 + idx] = qn[idx];
        wsV[512 + idx] = bs[idx] + us[idx];  wsV[640 + idx] = qs[idx];
    }
}

// ---------------- fused attention-aggregation, one workgroup per batch row ----------------
template<int D>
__global__ __launch_bounds__(256)
void gat_agg_kernel(const float* __restrict__ X,           // (B, 50, D) f32
                    const unsigned char* __restrict__ Msk, // (B, 50) bool
                    const bf16_t* __restrict__ Wbf,        // (128, D) bf16, row-major
                    const float* __restrict__ biasq,       // [bias(128) | q(128)]
                    float* __restrict__ Out)               // (B, D)
{
    constexpr int LSTR = D + 8;                 // bf16 elems; 16B-aligned, bank-staggered rows

    __shared__ bf16_t lA[MPAD * LSTR];
    __shared__ bf16_t lW[HDIM * LSTR];
    __shared__ float  sScore[MPAD];
    __shared__ float  sBias[HDIM];
    __shared__ float  sQ[HDIM];

    const int b    = blockIdx.x;
    const int tid  = threadIdx.x;
    const int lane = tid & 31;
    const int wave = tid >> 5;

    // ---- stage 0a: stage W (bf16) into LDS — async DMA path if available ----
    constexpr int WCHUNKS = HDIM * D / 8;       // 16-byte chunks
#if __has_builtin(__builtin_amdgcn_global_load_async_to_lds_b128)
    for (int i = tid; i < WCHUNKS; i += 256) {
        int row = (i * 8) / D, col = (i * 8) % D;
        __builtin_amdgcn_global_load_async_to_lds_b128(
            (v4i_gnu __attribute__((address_space(1)))*)(Wbf + i * 8),
            (v4i_gnu __attribute__((address_space(3)))*)&lW[row * LSTR + col],
            0, 0);
    }
#else
    for (int i = tid; i < WCHUNKS; i += 256) {
        int row = (i * 8) / D, col = (i * 8) % D;
        *(bf16x8*)(&lW[row * LSTR + col]) = *(const bf16x8*)(Wbf + i * 8);
    }
#endif

    // ---- stage 0b: stage x (f32 -> bf16) into LDS ----
    const float4* Xb4 = (const float4*)(X + (size_t)b * NNEI * D);
    for (int i = tid; i < NNEI * D / 4; i += 256) {
        float4 v = Xb4[i];
        int row = (i * 4) / D, col = (i * 4) % D;
        bf16x4 t;
        t[0] = (bf16_t)v.x; t[1] = (bf16_t)v.y; t[2] = (bf16_t)v.z; t[3] = (bf16_t)v.w;
        *(bf16x4*)(&lA[row * LSTR + col]) = t;
    }
    for (int i = tid; i < (MPAD - NNEI) * (D / 4); i += 256) {
        int row = NNEI + i / (D / 4), col = (i % (D / 4)) * 4;
        bf16x4 z; z[0] = (bf16_t)0.f; z[1] = (bf16_t)0.f; z[2] = (bf16_t)0.f; z[3] = (bf16_t)0.f;
        *(bf16x4*)(&lA[row * LSTR + col]) = z;
    }
    if (tid < HDIM) { sBias[tid] = biasq[tid]; sQ[tid] = biasq[HDIM + tid]; }
    if (tid < MPAD) sScore[tid] = 0.0f;
#if __has_builtin(__builtin_amdgcn_global_load_async_to_lds_b128)
#if __has_builtin(__builtin_amdgcn_s_wait_asynccnt)
    __builtin_amdgcn_s_wait_asynccnt(0);
#else
    asm volatile("s_wait_asynccnt 0" ::: "memory");
#endif
#endif
    __syncthreads();

    // ---- stage 1: h = tanh(x Wt + bias); score += h . q (WMMA bf16) ----
    // Wave w owns H-columns [w*16, w*16+16). 16-bit fragment layout (ISA 7.12.2):
    //   lanes 0-15: K-chunk base 0; lanes 16-31: K-chunk base 8;
    //   vector elems 0..7 = K[kb..kb+7], elems 8..15 = K[kb+16..kb+23].
    const int ncol = wave * 16 + (lane & 15);
    const int kb   = (lane >> 4) * 8;
    const float qn = sQ[ncol];
    const float bn = sBias[ncol];

    f32x8 acc[4];
#pragma unroll
    for (int mt = 0; mt < 4; ++mt)
        acc[mt] = (f32x8){0.f, 0.f, 0.f, 0.f, 0.f, 0.f, 0.f, 0.f};

    const bf16_t* wRow = &lW[ncol * LSTR];
    for (int kk = 0; kk < D; kk += 32) {
        bf16x8 blo = *(const bf16x8*)(wRow + kk + kb);
        bf16x8 bhi = *(const bf16x8*)(wRow + kk + kb + 16);
        bf16x16 bfrag = __builtin_shufflevector(blo, bhi,
            0, 1, 2, 3, 4, 5, 6, 7, 8, 9, 10, 11, 12, 13, 14, 15);
#pragma unroll
        for (int mt = 0; mt < 4; ++mt) {
            const bf16_t* aRow = &lA[(mt * 16 + (lane & 15)) * LSTR + kk + kb];
            bf16x8 alo = *(const bf16x8*)(aRow);
            bf16x8 ahi = *(const bf16x8*)(aRow + 16);
            bf16x16 afrag = __builtin_shufflevector(alo, ahi,
                0, 1, 2, 3, 4, 5, 6, 7, 8, 9, 10, 11, 12, 13, 14, 15);
            acc[mt] = __builtin_amdgcn_wmma_f32_16x16x32_bf16(
                false, afrag, false, bfrag, (short)0, acc[mt], false, false);
        }
    }

    // C/D layout: VGPR r -> M = r (lanes 0-15) or r+8 (lanes 16-31); N = lane&15.
#pragma unroll
    for (int mt = 0; mt < 4; ++mt) {
#pragma unroll
        for (int r = 0; r < 8; ++r) {
            float c = TANH_F32(acc[mt][r] + bn) * qn;
            c += __shfl_xor(c, 1);
            c += __shfl_xor(c, 2);
            c += __shfl_xor(c, 4);
            c += __shfl_xor(c, 8);
            if ((lane & 15) == 0) {
                int M = mt * 16 + r + ((lane >> 4) << 3);
                atomicAdd(&sScore[M], c);   // ds_add_f32
            }
        }
    }
    __syncthreads();

    // ---- stage 2: masked softmax over 50 (wave 0) ----
    if (wave == 0) {
        const unsigned char* mb = Msk + (size_t)b * NNEI;
        int n0 = lane, n1 = lane + 32;
        float s0 = (mb[n0] != 0) ? sScore[n0] : -INFINITY;
        float s1 = (n1 < NNEI && mb[n1] != 0) ? sScore[n1] : -INFINITY;
        float m = fmaxf(s0, s1);
#pragma unroll
        for (int off = 16; off > 0; off >>= 1) m = fmaxf(m, __shfl_xor(m, off));
        float e0 = __expf(s0 - m);
        float e1 = __expf(s1 - m);
        float s = e0 + e1;
#pragma unroll
        for (int off = 16; off > 0; off >>= 1) s += __shfl_xor(s, off);
        float inv = 1.0f / s;
        sScore[n0] = e0 * inv;
        if (n1 < NNEI) sScore[n1] = e1 * inv;
    }
    __syncthreads();

    // ---- stage 3: out[d] = sum_n attn[n] * x[n][d] ----
    for (int d = tid; d < D; d += 256) {
        float a = 0.0f;
#pragma unroll 5
        for (int n = 0; n < NNEI; ++n)
            a += sScore[n] * (float)lA[n * LSTR + d];
        Out[(size_t)b * D + d] = a;
    }
}

extern "C" void kernel_launch(void* const* d_in, const int* in_sizes, int n_in,
                              void* d_out, int out_size, void* d_ws, size_t ws_size,
                              hipStream_t stream) {
    const float* Xu  = (const float*)d_in[0];
    const float* Xn  = (const float*)d_in[1];
    const float* Xs  = (const float*)d_in[2];
    const unsigned char* Mu = (const unsigned char*)d_in[3];
    const unsigned char* Mn = (const unsigned char*)d_in[4];
    const float* Wu = (const float*)d_in[5];
    const float* bu = (const float*)d_in[6];
    const float* uu = (const float*)d_in[7];
    const float* qu = (const float*)d_in[8];
    const float* Wn = (const float*)d_in[9];
    const float* bn = (const float*)d_in[10];
    const float* un = (const float*)d_in[11];
    const float* qn = (const float*)d_in[12];
    const float* Ws = (const float*)d_in[13];
    const float* bs = (const float*)d_in[14];
    const float* us = (const float*)d_in[15];
    const float* qs = (const float*)d_in[16];

    const int B = in_sizes[0] / (NNEI * 128);
    float* out = (float*)d_out;
    float* out_users = out;
    float* out_news  = out + (size_t)B * 128;
    float* out_sem   = out + (size_t)B * 256;

    bf16_t* wsW = (bf16_t*)d_ws;
    float*  wsV = (float*)((char*)d_ws + (size_t)WS_W_ELEMS * sizeof(bf16_t));

    prep_weights<<<dim3((WS_W_ELEMS + 255) / 256), dim3(256), 0, stream>>>(
        Wu, bu, uu, qu, Wn, bn, un, qn, Ws, bs, us, qs, wsW, wsV);

    dim3 grid(B), block(256);
    gat_agg_kernel<128><<<grid, block, 0, stream>>>(Xu, Mu, wsW,          wsV,       out_users);
    gat_agg_kernel<128><<<grid, block, 0, stream>>>(Xn, Mn, wsW + 16384,  wsV + 256, out_news);
    gat_agg_kernel<384><<<grid, block, 0, stream>>>(Xs, Mn, wsW + 32768,  wsV + 512, out_sem);
}
